// PowerFlowGAT_61177514164373
// MI455X (gfx1250) — matrix operand order repaired and checked by reading, not compiled, not run on hardware
//
#include <hip/hip_runtime.h>
#include <math.h>

#define NNODE 10000
#define NEDGE 160000
#define HID 256
#define HEADS 8
#define CH 32
#define TLAYERS 5

typedef __attribute__((ext_vector_type(16))) __bf16 v16bf;
typedef __attribute__((ext_vector_type(8)))  __bf16 v8bf;
typedef __attribute__((ext_vector_type(8)))  float  v8f;

// ---------- float <-> order-preserving uint key (for atomic segment-max) ----
__device__ __forceinline__ unsigned f2key(float f) {
  unsigned u = __float_as_uint(f);
  return (u & 0x80000000u) ? ~u : (u | 0x80000000u);
}
__device__ __forceinline__ float key2f(unsigned k) {
  unsigned u = (k & 0x80000000u) ? (k & 0x7fffffffu) : ~k;
  return __uint_as_float(u);   // key==0 decodes to NaN -> treated as "empty"
}

// ---------------------------------------------------------------- utilities
__global__ void zero_f32(float* p, int n) {
  int i = blockIdx.x * blockDim.x + threadIdx.x;
  if (i < n) p[i] = 0.f;
}

// h = relu(x @ lift_W + lift_b)   [N,10] @ [10,256]
__global__ void lift_kernel(const float* __restrict__ x, const float* __restrict__ W,
                            const float* __restrict__ b, float* __restrict__ hf,
                            __bf16* __restrict__ hb) {
  int i = blockIdx.x * blockDim.x + threadIdx.x;
  if (i >= NNODE * HID) return;
  int row = i >> 8, col = i & 255;
  float acc = b[col];
#pragma unroll
  for (int d = 0; d < 10; ++d) acc += x[row * 10 + d] * W[d * HID + col];
  acc = fmaxf(acc, 0.f);
  hf[i] = acc;
  hb[i] = (__bf16)acc;
}

// per-column mean / 1/(std+1e-8) over the E edges (population std)
__global__ void edge_stats(const float* __restrict__ ea, float* __restrict__ st) {
  __shared__ float s1[256], s2[256];
  int col = blockIdx.x, tid = threadIdx.x;
  float a = 0.f, b = 0.f;
  for (int i = tid; i < NEDGE; i += 256) {
    float v = ea[i * 9 + col];
    a += v; b += v * v;
  }
  s1[tid] = a; s2[tid] = b;
  __syncthreads();
  for (int o = 128; o > 0; o >>= 1) {
    if (tid < o) { s1[tid] += s1[tid + o]; s2[tid] += s2[tid + o]; }
    __syncthreads();
  }
  if (tid == 0) {
    float mu = s1[0] / (float)NEDGE;
    float var = s2[0] / (float)NEDGE - mu * mu;
    st[col * 2] = mu;
    st[col * 2 + 1] = 1.f / (sqrtf(fmaxf(var, 0.f)) + 1e-8f);
  }
}

__global__ void edge_norm(const float* __restrict__ ea, const float* __restrict__ st,
                          float* __restrict__ ean) {
  int i = blockIdx.x * blockDim.x + threadIdx.x;
  if (i >= NEDGE * 9) return;
  int col = i % 9;
  ean[i] = (ea[i] - st[col * 2]) * st[col * 2 + 1];
}

// Pack row-major f32 [K=256, Ncols=256] weight into bf16 B-fragment layout:
// P[nt][kt][lane][e] where n = nt*16 + (lane&15), k = kt*32 + (lane>>4)*16 + e
__global__ void pack_b(const float* __restrict__ W, __bf16* __restrict__ P) {
  int idx = blockIdx.x * blockDim.x + threadIdx.x;   // 16*8*32*16 = 65536
  if (idx >= 65536) return;
  int e    = idx & 15;
  int lane = (idx >> 4) & 31;
  int kt   = (idx >> 9) & 7;
  int nt   = idx >> 12;
  int n = nt * 16 + (lane & 15);
  int k = kt * 32 + ((lane >> 4) << 4) + e;
  P[idx] = (__bf16)W[k * HID + n];
}

// D[M,256] = A[M,256](bf16) @ Bpacked + bias ; one wave per 16x16 tile, K=256
__global__ void gemm_bf16_wmma(const __bf16* __restrict__ A, const __bf16* __restrict__ Bp,
                               const float* __restrict__ bias, float* __restrict__ D) {
  int mt = blockIdx.x;          // 0..624
  int nt = blockIdx.y;          // 0..15
  int lane = threadIdx.x;       // 0..31
  int half = lane >> 4;
  int l16  = lane & 15;
  const __bf16* Arow = A + (size_t)(mt * 16 + l16) * HID + half * 16;
  const __bf16* Bl   = Bp + ((size_t)nt * 8) * 512 + lane * 16;
  v8f acc = {};
#pragma unroll
  for (int kt = 0; kt < 8; ++kt) {
    v8bf alo = *(const v8bf*)(Arow + kt * 32);        // K offsets +0..7
    v8bf ahi = *(const v8bf*)(Arow + kt * 32 + 8);    // K offsets +8..15
    v16bf a;
#pragma unroll
    for (int q = 0; q < 8; ++q) { a[q] = alo[q]; a[q + 8] = ahi[q]; }
    v16bf b = *(const v16bf*)(Bl + kt * 512);
    acc = __builtin_amdgcn_wmma_f32_16x16x32_bf16(false, a, false, b,
                                                  (short)0, acc, false, false);
  }
  int col  = nt * 16 + l16;
  float bb = bias[col];
  int row0 = mt * 16 + half * 8;
#pragma unroll
  for (int q = 0; q < 8; ++q)
    D[(size_t)(row0 + q) * HID + col] = acc[q] + bb;
}

// logits[e,h] = sum_c lrelu(xl[src]+xr[dst]+ (ea@We))*att ; xe computed on the fly
__global__ void att_logits(const float* __restrict__ xl, const float* __restrict__ xr,
                           const float* __restrict__ ean, const int* __restrict__ ei,
                           const float* __restrict__ We, const float* __restrict__ att,
                           float* __restrict__ logits, unsigned* __restrict__ mkey) {
  int gid = blockIdx.x * blockDim.x + threadIdx.x;
  if (gid >= NEDGE * HEADS) return;
  int e = gid >> 3, h = gid & 7;
  int src = ei[e], dst = ei[NEDGE + e];
  float ev[9];
#pragma unroll
  for (int d = 0; d < 9; ++d) ev[d] = ean[e * 9 + d];
  const float* xls = xl + (size_t)src * HID + h * CH;
  const float* xrd = xr + (size_t)dst * HID + h * CH;
  const float* ah  = att + h * CH;
  float lg = 0.f;
#pragma unroll 4
  for (int c = 0; c < CH; ++c) {
    int j = h * CH + c;
    float xe = 0.f;
#pragma unroll
    for (int d = 0; d < 9; ++d) xe += ev[d] * We[d * HID + j];
    float v = xls[c] + xrd[c] + xe;
    v = (v > 0.f) ? v : 0.2f * v;     // leaky_relu, slope 0.2
    lg += v * ah[c];
  }
  logits[gid] = lg;
  atomicMax(mkey + dst * HEADS + h, f2key(lg));
}

// z = exp(logit - m[dst]); denom[dst] += z (in-place overwrite of logits)
__global__ void att_exp(const int* __restrict__ ei, float* __restrict__ zbuf,
                        const unsigned* __restrict__ mkey, float* __restrict__ denom) {
  int gid = blockIdx.x * blockDim.x + threadIdx.x;
  if (gid >= NEDGE * HEADS) return;
  int e = gid >> 3, h = gid & 7;
  int dst = ei[NEDGE + e];
  float m = key2f(mkey[dst * HEADS + h]);
  if (!isfinite(m)) m = 0.f;
  float z = expf(zbuf[gid] - m);
  zbuf[gid] = z;
  atomicAdd(denom + dst * HEADS + h, z);
}

// agg[dst,j] += alpha * xl[src,j] ; one block of 256 threads per edge
__global__ void att_agg(const int* __restrict__ ei, const float* __restrict__ zbuf,
                        const float* __restrict__ denom, const float* __restrict__ xl,
                        float* __restrict__ agg) {
  int e = blockIdx.x;
  int j = threadIdx.x;
  int h = j >> 5;
  int src = ei[e], dst = ei[NEDGE + e];
  float alpha = zbuf[e * HEADS + h] / (denom[dst * HEADS + h] + 1e-16f);
  atomicAdd(agg + (size_t)dst * HID + j, alpha * xl[(size_t)src * HID + j]);
}

// BatchNorm column stats over N rows (optionally adding a per-col bias first)
__global__ void col_stats(const float* __restrict__ X, const float* __restrict__ bias,
                          float* __restrict__ st) {
  __shared__ float s1[256], s2[256];
  int col = blockIdx.x, tid = threadIdx.x;
  float bb = bias ? bias[col] : 0.f;
  float a = 0.f, b = 0.f;
  for (int i = tid; i < NNODE; i += 256) {
    float v = X[(size_t)i * HID + col] + bb;
    a += v; b += v * v;
  }
  s1[tid] = a; s2[tid] = b;
  __syncthreads();
  for (int o = 128; o > 0; o >>= 1) {
    if (tid < o) { s1[tid] += s1[tid + o]; s2[tid] += s2[tid + o]; }
    __syncthreads();
  }
  if (tid == 0) {
    float mu = s1[0] / (float)NNODE;
    float var = s2[0] / (float)NNODE - mu * mu;
    st[col * 2] = mu;
    st[col * 2 + 1] = rsqrtf(fmaxf(var, 0.f) + 1e-5f);
  }
}

// h = act(gamma*(X+bias-mu)*rstd + beta) ; elu=1 -> ELU, elu=0 -> ReLU
__global__ void bn_act(const float* __restrict__ X, const float* __restrict__ bias,
                       const float* __restrict__ st, const float* __restrict__ gamma,
                       const float* __restrict__ beta, float* __restrict__ hf,
                       __bf16* __restrict__ hb, int elu) {
  int i = blockIdx.x * blockDim.x + threadIdx.x;
  if (i >= NNODE * HID) return;
  int col = i & 255;
  float v = X[i] + (bias ? bias[col] : 0.f);
  v = gamma[col] * (v - st[col * 2]) * st[col * 2 + 1] + beta[col];
  v = elu ? (v > 0.f ? v : expm1f(v)) : fmaxf(v, 0.f);
  hf[i] = v;
  hb[i] = (__bf16)v;
}

// out[N,3] = h @ p3_W + p3_b
__global__ void proj_out(const float* __restrict__ hf, const float* __restrict__ W,
                         const float* __restrict__ b, float* __restrict__ out) {
  int gid = blockIdx.x * blockDim.x + threadIdx.x;
  if (gid >= NNODE * 3) return;
  int i = gid / 3, o = gid % 3;
  float acc = b[o];
  for (int j = 0; j < HID; ++j) acc += hf[(size_t)i * HID + j] * W[j * 3 + o];
  out[gid] = acc;
}

extern "C" void kernel_launch(void* const* d_in, const int* in_sizes, int n_in,
                              void* d_out, int out_size, void* d_ws, size_t ws_size,
                              hipStream_t stream) {
  (void)in_sizes; (void)n_in; (void)out_size; (void)ws_size;
  const float* x         = (const float*)d_in[0];
  const float* edge_attr = (const float*)d_in[1];
  const int*   ei        = (const int*)d_in[2];
  const float* lift_W    = (const float*)d_in[3];
  const float* lift_b    = (const float*)d_in[4];
  const float* Wl        = (const float*)d_in[5];
  const float* bl        = (const float*)d_in[6];
  const float* Wr        = (const float*)d_in[7];
  const float* br        = (const float*)d_in[8];
  const float* We        = (const float*)d_in[9];
  const float* att       = (const float*)d_in[10];
  const float* conv_bias = (const float*)d_in[11];
  const float* bn_gamma  = (const float*)d_in[12];
  const float* bn_beta   = (const float*)d_in[13];
  const float* p1_W = (const float*)d_in[14];
  const float* p1_b = (const float*)d_in[15];
  const float* pbn1_g = (const float*)d_in[16];
  const float* pbn1_b = (const float*)d_in[17];
  const float* p2_W = (const float*)d_in[18];
  const float* p2_b = (const float*)d_in[19];
  const float* pbn2_g = (const float*)d_in[20];
  const float* pbn2_b = (const float*)d_in[21];
  const float* p3_W = (const float*)d_in[22];
  const float* p3_b = (const float*)d_in[23];

  char* ws = (char*)d_ws;
  size_t off = 0;
  auto alloc = [&](size_t bytes) -> char* {
    char* p = ws + off;
    off = (off + bytes + 255) & ~(size_t)255;
    return p;
  };
  float*    ean   = (float*)   alloc((size_t)NEDGE * 9 * 4);
  __bf16*   h_bf  = (__bf16*)  alloc((size_t)NNODE * HID * 2);
  float*    h_f   = (float*)   alloc((size_t)NNODE * HID * 4);
  float*    xl    = (float*)   alloc((size_t)NNODE * HID * 4);
  float*    xr    = (float*)   alloc((size_t)NNODE * HID * 4);
  float*    zbuf  = (float*)   alloc((size_t)NEDGE * HEADS * 4);
  unsigned* mkey  = (unsigned*)alloc((size_t)NNODE * HEADS * 4);
  float*    denom = (float*)   alloc((size_t)NNODE * HEADS * 4);
  float*    agg   = (float*)   alloc((size_t)NNODE * HID * 4);
  __bf16*   packW = (__bf16*)  alloc((size_t)12 * 65536 * 2);
  float*    estat = (float*)   alloc(9 * 2 * 4);
  float*    cstat = (float*)   alloc(HID * 2 * 4);

  // edge feature normalization (z-score per column)
  edge_stats<<<9, 256, 0, stream>>>(edge_attr, estat);
  edge_norm<<<(NEDGE * 9 + 255) / 256, 256, 0, stream>>>(edge_attr, estat, ean);
  // lift
  lift_kernel<<<(NNODE * HID + 255) / 256, 256, 0, stream>>>(x, lift_W, lift_b, h_f, h_bf);
  // pack all 256x256 weights into bf16 WMMA B-fragment layout
  for (int t = 0; t < TLAYERS; ++t) {
    pack_b<<<256, 256, 0, stream>>>(Wl + (size_t)t * HID * HID, packW + (size_t)t * 65536);
    pack_b<<<256, 256, 0, stream>>>(Wr + (size_t)t * HID * HID, packW + (size_t)(5 + t) * 65536);
  }
  pack_b<<<256, 256, 0, stream>>>(p1_W, packW + (size_t)10 * 65536);
  pack_b<<<256, 256, 0, stream>>>(p2_W, packW + (size_t)11 * 65536);

  dim3 ggrid(NNODE / 16, HID / 16);   // 625 x 16 tiles, one wave each
  for (int t = 0; t < TLAYERS; ++t) {
    gemm_bf16_wmma<<<ggrid, 32, 0, stream>>>(h_bf, packW + (size_t)t * 65536, bl + t * HID, xl);
    gemm_bf16_wmma<<<ggrid, 32, 0, stream>>>(h_bf, packW + (size_t)(5 + t) * 65536, br + t * HID, xr);
    zero_f32<<<(NNODE * HEADS + 255) / 256, 256, 0, stream>>>((float*)mkey, NNODE * HEADS);
    zero_f32<<<(NNODE * HEADS + 255) / 256, 256, 0, stream>>>(denom, NNODE * HEADS);
    zero_f32<<<(NNODE * HID + 255) / 256, 256, 0, stream>>>(agg, NNODE * HID);
    att_logits<<<(NEDGE * HEADS + 255) / 256, 256, 0, stream>>>(
        xl, xr, ean, ei, We + (size_t)t * 9 * HID, att + (size_t)t * HEADS * CH, zbuf, mkey);
    att_exp<<<(NEDGE * HEADS + 255) / 256, 256, 0, stream>>>(ei, zbuf, mkey, denom);
    att_agg<<<NEDGE, 256, 0, stream>>>(ei, zbuf, denom, xl, agg);
    col_stats<<<HID, 256, 0, stream>>>(agg, conv_bias + t * HID, cstat);
    bn_act<<<(NNODE * HID + 255) / 256, 256, 0, stream>>>(
        agg, conv_bias + t * HID, cstat, bn_gamma + t * HID, bn_beta + t * HID, h_f, h_bf, 1);
  }
  // projection MLP: Lin-BN-ReLU, Lin-BN-ReLU, Lin
  gemm_bf16_wmma<<<ggrid, 32, 0, stream>>>(h_bf, packW + (size_t)10 * 65536, p1_b, xl);
  col_stats<<<HID, 256, 0, stream>>>(xl, nullptr, cstat);
  bn_act<<<(NNODE * HID + 255) / 256, 256, 0, stream>>>(xl, nullptr, cstat, pbn1_g, pbn1_b, h_f, h_bf, 0);
  gemm_bf16_wmma<<<ggrid, 32, 0, stream>>>(h_bf, packW + (size_t)11 * 65536, p2_b, xl);
  col_stats<<<HID, 256, 0, stream>>>(xl, nullptr, cstat);
  bn_act<<<(NNODE * HID + 255) / 256, 256, 0, stream>>>(xl, nullptr, cstat, pbn2_g, pbn2_b, h_f, h_bf, 0);
  proj_out<<<(NNODE * 3 + 255) / 256, 256, 0, stream>>>(h_f, p3_W, p3_b, (float*)d_out);
}
